// cfpnHead_89378269430384
// MI455X (gfx1250) — matrix operand order repaired and checked by reading, not compiled, not run on hardware
//
#include <hip/hip_runtime.h>
#include <cstdint>
#include <cstddef>

// ---------------------------------------------------------------------------
// Types for CDNA5 WMMA (gfx1250, wave32)
// ---------------------------------------------------------------------------
typedef __bf16 bf16;
typedef __attribute__((ext_vector_type(16))) __bf16 v16bf;
typedef __attribute__((ext_vector_type(8)))  __bf16 bf16x8;
typedef __attribute__((ext_vector_type(8)))  float  v8f;

static __device__ inline float b2f(bf16 x) { return (float)x; }
static __device__ inline bf16  f2b(float x) { return (bf16)x; }

static __device__ inline v16bf zero_v16() {
  v16bf z;
#pragma unroll
  for (int e = 0; e < 16; ++e) z[e] = f2b(0.0f);
  return z;
}

// ---------------------------------------------------------------------------
// Layout conversion kernels
// ---------------------------------------------------------------------------
__global__ void nchw2nhwc_kernel(const float* __restrict__ in, bf16* __restrict__ out,
                                 int Bn, int C, int H, int W) {
  size_t tot = (size_t)Bn * C * H * W;
  size_t i = (size_t)blockIdx.x * blockDim.x + threadIdx.x;
  if (i >= tot) return;
  int w = (int)(i % W); size_t r = i / W;
  int h = (int)(r % H); r /= H;
  int c = (int)(r % C); int b = (int)(r / C);
  out[(((size_t)b * H + h) * W + w) * C + c] = f2b(in[i]);
}

// OIHW (f32) -> (O, KH*KW, I) (bf16)
__global__ void wcvt_kernel(const float* __restrict__ in, bf16* __restrict__ out,
                            int O, int T, int I) {
  size_t tot = (size_t)O * T * I;
  size_t i = (size_t)blockIdx.x * blockDim.x + threadIdx.x;
  if (i >= tot) return;
  int ii = (int)(i % I); size_t r = i / I;
  int t = (int)(r % T); int o = (int)(r / T);
  out[i] = f2b(in[((size_t)o * I + ii) * T + t]);
}

__global__ void copy_ch_kernel(const bf16* __restrict__ src, bf16* __restrict__ dst,
                               size_t P, int Cs, int Cd, int coff) {
  size_t tot = P * (size_t)Cs;
  size_t i = (size_t)blockIdx.x * blockDim.x + threadIdx.x;
  if (i >= tot) return;
  size_t p = i / Cs; int c = (int)(i % Cs);
  dst[p * (size_t)Cd + coff + c] = src[i];
}

// ---------------------------------------------------------------------------
// Implicit-GEMM conv with WMMA (bf16 in, f32 accum)
//   X  : (Bn,H,W,Cin) NHWC bf16 ; H*W and W are powers of two (lhw/lw shifts)
//   Wt : (Cout, ks*ks, Cin) bf16
//   act: 0 none, 1 relu, 2 sigmoid. scale/bias optional (BN fold).
//   Yb : NHWC bf16 out; if Yf != nullptr, store NCHW f32 instead.
// Block = 256 thr = 8 waves, tiled 2(M)x4(N); each wave owns a 16x32 tile:
// one A fragment feeds two WMMAs (two accumulators), doubling compute per
// A byte pulled from HBM (weights stream from the 192MB L2).
// ---------------------------------------------------------------------------
__global__ __launch_bounds__(256) void conv_wmma_kernel(
    const bf16* __restrict__ X, const bf16* __restrict__ Wt,
    const float* __restrict__ scale, const float* __restrict__ bias,
    bf16* __restrict__ Yb, float* __restrict__ Yf,
    int Bn, int H, int W, int Cin, int Cout, int ks, int dil, int act,
    int lhw, int lw) {
  const int M = Bn * (H * W);
  const int lane = threadIdx.x & 31;
  const int wid  = threadIdx.x >> 5;
  const int m0 = blockIdx.x * 32 + (wid & 1) * 16;
  const int n0 = blockIdx.y * 128 + (wid >> 1) * 32;
  const int lm = lane & 15;
  const int lk = lane >> 4;
  const int mRow  = m0 + lm;           // A row this lane supplies
  const int nCol0 = n0 + lm;           // B column (sub-tile 0)
  const int nCol1 = n0 + 16 + lm;      // B column (sub-tile 1)
  const bool rowOK = mRow < M;
  const bool colOK0 = nCol0 < Cout;
  const bool colOK1 = nCol1 < Cout;
  const int hwMask = (1 << lhw) - 1;
  const int wMask  = (1 << lw) - 1;

  int nn = 0, oh = 0, ow = 0;
  if (rowOK) {
    nn = mRow >> lhw;
    int r = mRow & hwMask;
    oh = r >> lw;
    ow = r & wMask;
  }

  const int taps = ks * ks;
  const int pad = (ks == 3) ? dil : 0;
  v8f acc0 = {0.f, 0.f, 0.f, 0.f, 0.f, 0.f, 0.f, 0.f};
  v8f acc1 = {0.f, 0.f, 0.f, 0.f, 0.f, 0.f, 0.f, 0.f};

  for (int t = 0; t < taps; ++t) {
    const int ky = t / ks, kx = t - ky * ks;
    const int ih = oh + ky * dil - pad;
    const int iw = ow + kx * dil - pad;
    const bool v = rowOK && ((unsigned)ih < (unsigned)H) && ((unsigned)iw < (unsigned)W);
    const bf16* arow = X + (((size_t)nn * H + (v ? ih : 0)) * W + (v ? iw : 0)) * Cin;
    const bf16* brow0 = Wt + ((size_t)(colOK0 ? nCol0 : 0) * taps + t) * Cin;
    const bf16* brow1 = Wt + ((size_t)(colOK1 ? nCol1 : 0) * taps + t) * Cin;
    if (v) __builtin_prefetch(arow, 0, 0);   // global_prefetch_b8
    for (int k0 = 0; k0 < Cin; k0 += 32) {
      v16bf a = zero_v16();
      v16bf bm0 = zero_v16();
      v16bf bm1 = zero_v16();
      if (v) {
        const int ka = k0 + lk * 16;
        bf16x8 lo = *(const bf16x8*)(arow + ka);
        bf16x8 hi = *(const bf16x8*)(arow + ka + 8);
#pragma unroll
        for (int e = 0; e < 8; ++e) { a[e] = lo[e]; a[e + 8] = hi[e]; }
      }
      const int kb = k0 + lk * 16;
      if (colOK0) {
        bf16x8 lo = *(const bf16x8*)(brow0 + kb);
        bf16x8 hi = *(const bf16x8*)(brow0 + kb + 8);
#pragma unroll
        for (int e = 0; e < 8; ++e) { bm0[e] = lo[e]; bm0[e + 8] = hi[e]; }
      }
      if (colOK1) {
        bf16x8 lo = *(const bf16x8*)(brow1 + kb);
        bf16x8 hi = *(const bf16x8*)(brow1 + kb + 8);
#pragma unroll
        for (int e = 0; e < 8; ++e) { bm1[e] = lo[e]; bm1[e + 8] = hi[e]; }
      }
      acc0 = __builtin_amdgcn_wmma_f32_16x16x32_bf16(false, a, false, bm0, (short)0, acc0,
                                                     false, false);
      acc1 = __builtin_amdgcn_wmma_f32_16x16x32_bf16(false, a, false, bm1, (short)0, acc1,
                                                     false, false);
    }
  }

#pragma unroll
  for (int j = 0; j < 2; ++j) {
    const int c = n0 + j * 16 + lm;
    if (c >= Cout) continue;
    const float s  = scale ? scale[c] : 1.f;
    const float bb = bias ? bias[c] : 0.f;
#pragma unroll
    for (int r = 0; r < 8; ++r) {
      const int m = m0 + r + 8 * lk;
      if (m >= M) continue;
      float v = (j == 0 ? acc0[r] : acc1[r]) * s + bb;
      if (act == 1) v = v > 0.f ? v : 0.f;
      else if (act == 2) v = 1.f / (1.f + expf(-v));
      if (Yf) {
        int n2 = m >> lhw; int rr = m & hwMask; int h2 = rr >> lw; int w2 = rr & wMask;
        Yf[(((size_t)n2 * Cout + c) * H + h2) * W + w2] = v;
      } else {
        Yb[(size_t)m * Cout + c] = f2b(v);
      }
    }
  }
}

// ---------------------------------------------------------------------------
// Batched GEMM with WMMA: C[b] = A[b] (MxK) * B[b]
//   bkn==0: B stored (N,K) row-major (contiguous K per column) -> fast path
//   bkn==1: B stored (K,N) row-major (strided gather)
// Output f32 (Cf) or bf16 (Cb), row-major (M,N). 16x32 tile per wave.
// ---------------------------------------------------------------------------
__global__ __launch_bounds__(256) void gemm_wmma_kernel(
    const bf16* __restrict__ A, const bf16* __restrict__ Bm,
    float* __restrict__ Cf, bf16* __restrict__ Cb,
    int M, int N, int K, int bkn,
    unsigned long long sA, unsigned long long sB, unsigned long long sC) {
  const int bz = blockIdx.z;
  A  += (size_t)bz * sA;
  Bm += (size_t)bz * sB;
  const int lane = threadIdx.x & 31;
  const int wid  = threadIdx.x >> 5;
  const int m0 = blockIdx.x * 32 + (wid & 1) * 16;
  const int n0 = blockIdx.y * 128 + (wid >> 1) * 32;
  const int lm = lane & 15;
  const int lk = lane >> 4;
  const int mRow  = m0 + lm;
  const int nCol0 = n0 + lm;
  const int nCol1 = n0 + 16 + lm;
  const bool rowOK = mRow < M;
  const bool colOK0 = nCol0 < N;
  const bool colOK1 = nCol1 < N;
  const bf16* arow = A + (size_t)(rowOK ? mRow : 0) * K;
  const bf16* brow0 = Bm + (size_t)(colOK0 ? nCol0 : 0) * K;
  const bf16* brow1 = Bm + (size_t)(colOK1 ? nCol1 : 0) * K;

  v8f acc0 = {0.f, 0.f, 0.f, 0.f, 0.f, 0.f, 0.f, 0.f};
  v8f acc1 = {0.f, 0.f, 0.f, 0.f, 0.f, 0.f, 0.f, 0.f};
  for (int k0 = 0; k0 < K; k0 += 32) {
    v16bf a = zero_v16();
    v16bf bm0 = zero_v16();
    v16bf bm1 = zero_v16();
    if (rowOK) {
      const int ka = k0 + lk * 16;
      bf16x8 lo = *(const bf16x8*)(arow + ka);
      bf16x8 hi = *(const bf16x8*)(arow + ka + 8);
#pragma unroll
      for (int e = 0; e < 8; ++e) { a[e] = lo[e]; a[e + 8] = hi[e]; }
    }
    const int kb = k0 + lk * 16;
    if (!bkn) {
      if (colOK0) {
        bf16x8 lo = *(const bf16x8*)(brow0 + kb);
        bf16x8 hi = *(const bf16x8*)(brow0 + kb + 8);
#pragma unroll
        for (int e = 0; e < 8; ++e) { bm0[e] = lo[e]; bm0[e + 8] = hi[e]; }
      }
      if (colOK1) {
        bf16x8 lo = *(const bf16x8*)(brow1 + kb);
        bf16x8 hi = *(const bf16x8*)(brow1 + kb + 8);
#pragma unroll
        for (int e = 0; e < 8; ++e) { bm1[e] = lo[e]; bm1[e + 8] = hi[e]; }
      }
    } else {
      if (colOK0) {
#pragma unroll
        for (int e = 0; e < 16; ++e) bm0[e] = Bm[(size_t)(kb + e) * N + nCol0];
      }
      if (colOK1) {
#pragma unroll
        for (int e = 0; e < 16; ++e) bm1[e] = Bm[(size_t)(kb + e) * N + nCol1];
      }
    }
    acc0 = __builtin_amdgcn_wmma_f32_16x16x32_bf16(false, a, false, bm0, (short)0, acc0,
                                                   false, false);
    acc1 = __builtin_amdgcn_wmma_f32_16x16x32_bf16(false, a, false, bm1, (short)0, acc1,
                                                   false, false);
  }
#pragma unroll
  for (int j = 0; j < 2; ++j) {
    const int c = n0 + j * 16 + lm;
    if (c >= N) continue;
#pragma unroll
    for (int r = 0; r < 8; ++r) {
      const int m = m0 + r + 8 * lk;
      if (m >= M) continue;
      float v = (j == 0 ? acc0[r] : acc1[r]);
      if (Cf) Cf[(size_t)bz * sC + (size_t)m * N + c] = v;
      else    Cb[(size_t)bz * sC + (size_t)m * N + c] = f2b(v);
    }
  }
}

// ---------------------------------------------------------------------------
// Elementwise / pooling / resize / reductions
// ---------------------------------------------------------------------------
__global__ void add_relu_kernel(const bf16* __restrict__ a, const bf16* __restrict__ b,
                                bf16* __restrict__ y, size_t tot) {
  size_t i = (size_t)blockIdx.x * blockDim.x + threadIdx.x;
  if (i >= tot) return;
  float v = b2f(a[i]) + b2f(b[i]);
  y[i] = f2b(v > 0.f ? v : 0.f);
}

__global__ void maxpool2_kernel(const bf16* __restrict__ X, bf16* __restrict__ Y,
                                int Bn, int H, int W, int C) {
  const int OH = H / 2, OW = W / 2;
  size_t tot = (size_t)Bn * OH * OW * C;
  size_t i = (size_t)blockIdx.x * blockDim.x + threadIdx.x;
  if (i >= tot) return;
  int c = (int)(i % C); size_t r = i / C;
  int ow = (int)(r % OW); r /= OW;
  int oh = (int)(r % OH); int b = (int)(r / OH);
  const bf16* p = X + (((size_t)b * H + oh * 2) * W + ow * 2) * C + c;
  size_t rs = (size_t)W * C;
  float v = fmaxf(fmaxf(b2f(p[0]), b2f(p[C])), fmaxf(b2f(p[rs]), b2f(p[rs + C])));
  Y[i] = f2b(v);
}

// align-corners bilinear, NHWC, writes into (Cd, coff) slice of dst
__global__ void resize_ac_kernel(const bf16* __restrict__ X, bf16* __restrict__ Y,
                                 int Bn, int IH, int IW, int C, int OH, int OW,
                                 int Cd, int coff) {
  size_t tot = (size_t)Bn * OH * OW * C;
  size_t i = (size_t)blockIdx.x * blockDim.x + threadIdx.x;
  if (i >= tot) return;
  int c = (int)(i % C); size_t r = i / C;
  int ow = (int)(r % OW); r /= OW;
  int oh = (int)(r % OH); int b = (int)(r / OH);
  float sy = (OH > 1) ? (float)(IH - 1) / (float)(OH - 1) : 0.f;
  float sx = (OW > 1) ? (float)(IW - 1) / (float)(OW - 1) : 0.f;
  float ty = oh * sy;
  int y0 = (int)floorf(ty); y0 = y0 < 0 ? 0 : (y0 > IH - 1 ? IH - 1 : y0);
  int y1 = y0 + 1 > IH - 1 ? IH - 1 : y0 + 1;
  float wy = ty - (float)y0;
  float tx = ow * sx;
  int x0 = (int)floorf(tx); x0 = x0 < 0 ? 0 : (x0 > IW - 1 ? IW - 1 : x0);
  int x1 = x0 + 1 > IW - 1 ? IW - 1 : x0 + 1;
  float wx = tx - (float)x0;
  auto at = [&](int yy, int xx) -> float {
    return b2f(X[(((size_t)b * IH + yy) * IW + xx) * C + c]);
  };
  float v = (at(y0, x0) * (1.f - wy) + at(y1, x0) * wy) * (1.f - wx) +
            (at(y0, x1) * (1.f - wy) + at(y1, x1) * wy) * wx;
  Y[(((size_t)b * OH + oh) * OW + ow) * Cd + coff + c] = f2b(v);
}

// mean over H,W from the original NCHW f32 tensor (contiguous per (n,c))
__global__ void gap_mean_kernel(const float* __restrict__ X, bf16* __restrict__ O,
                                int BC, int HW) {
  int i = blockIdx.x * blockDim.x + threadIdx.x;
  if (i >= BC) return;
  const float* p = X + (size_t)i * HW;
  float s = 0.f;
  for (int j = 0; j < HW; ++j) s += p[j];
  O[i] = f2b(s / (float)HW);
}

// x *= (1 + se[b,c]) over NHWC
__global__ void scale_se_kernel(bf16* __restrict__ X, const bf16* __restrict__ SE,
                                size_t tot, int P, int C) {
  size_t i = (size_t)blockIdx.x * blockDim.x + threadIdx.x;
  if (i >= tot) return;
  int c = (int)(i % C);
  int b = (int)(i / ((size_t)P * C));
  X[i] = f2b(b2f(X[i]) * (1.f + b2f(SE[(size_t)b * C + c])));
}

__global__ __launch_bounds__(256) void softmax_rows_kernel(const float* __restrict__ S,
                                                           bf16* __restrict__ O, int L) {
  __shared__ float red[256];
  size_t row = blockIdx.x;
  const float* s = S + row * (size_t)L;
  bf16* o = O + row * (size_t)L;
  int tid = threadIdx.x;
  float mx = -3.4e38f;
  for (int i = tid; i < L; i += 256) mx = fmaxf(mx, s[i]);
  red[tid] = mx; __syncthreads();
  for (int st = 128; st > 0; st >>= 1) { if (tid < st) red[tid] = fmaxf(red[tid], red[tid + st]); __syncthreads(); }
  mx = red[0]; __syncthreads();
  float sum = 0.f;
  for (int i = tid; i < L; i += 256) sum += expf(s[i] - mx);
  red[tid] = sum; __syncthreads();
  for (int st = 128; st > 0; st >>= 1) { if (tid < st) red[tid] += red[tid + st]; __syncthreads(); }
  float inv = 1.f / red[0];
  for (int i = tid; i < L; i += 256) o[i] = f2b(expf(s[i] - mx) * inv);
}

__global__ void pam_mix_kernel(const bf16* __restrict__ core, const bf16* __restrict__ prev,
                               const bf16* __restrict__ g, bf16* __restrict__ out,
                               size_t tot, int C) {
  size_t i = (size_t)blockIdx.x * blockDim.x + threadIdx.x;
  if (i >= tot) return;
  size_t pix = i / C;
  float gv = b2f(g[pix]);
  out[i] = f2b((1.f - gv) * b2f(core[i]) + gv * b2f(prev[i]));
}

// ---------------------------------------------------------------------------
// local_up2 gather attention
// ---------------------------------------------------------------------------
static __device__ inline void lu2_geom(int p, int WD, int HS, int WS,
                                       float sh, float sw, int* idx, float* coef) {
  int i = p / WD, j = p % WD;
  float sx = ((float)i + 0.5f) * sh - 0.5f;
  float sy = ((float)j + 0.5f) * sw - 0.5f;
  int x0 = (int)floorf(sx), y0 = (int)floorf(sy);
  int x1 = (x0 + 1 < HS - 1) ? x0 + 1 : HS - 1;
  int y1 = (y0 + 1 < WS - 1) ? y0 + 1 : WS - 1;
  int x00 = x0 > 0 ? x0 : 0;
  int y00 = y0 > 0 ? y0 : 0;
  idx[0] = y00 * HS + x00; idx[1] = y1 * HS + x00;
  idx[2] = y00 * HS + x1;  idx[3] = y1 * HS + x1;
  float norm = fmaxf((float)((x1 - x0) * (y1 - y0)), 1.f);
  coef[0] = (sx - x0) * (sy - y0) / norm;
  coef[1] = (sx - x0) * (y1 - sy) / norm;
  coef[2] = (x1 - sx) * (sy - y0) / norm;
  coef[3] = (x1 - sx) * (y1 - sy) / norm;
}

__global__ void lu2_att_kernel(const bf16* __restrict__ Q, const bf16* __restrict__ Kb,
                               float* __restrict__ Att, int Bn, int HD, int WD,
                               int HS, int WS, int CK) {
  const int P = HD * WD;
  size_t t = (size_t)blockIdx.x * blockDim.x + threadIdx.x;
  if (t >= (size_t)Bn * P) return;
  int b = (int)(t / P), p = (int)(t % P);
  int idx[4]; float coef[4];
  lu2_geom(p, WD, HS, WS, (float)HS / HD, (float)WS / WD, idx, coef);
  const bf16* q = Q + ((size_t)b * P + p) * CK;
  float lg[4];
  for (int f = 0; f < 4; ++f) {
    const bf16* kr = Kb + ((size_t)b * HS * WS + idx[f]) * CK;
    float d = 0.f;
    for (int k = 0; k < CK; ++k) d += b2f(q[k]) * b2f(kr[k]);
    lg[f] = d;
  }
  float mx = fmaxf(fmaxf(lg[0], lg[1]), fmaxf(lg[2], lg[3]));
  float e0 = expf(lg[0] - mx), e1 = expf(lg[1] - mx);
  float e2 = expf(lg[2] - mx), e3 = expf(lg[3] - mx);
  float inv = 1.f / (e0 + e1 + e2 + e3);
  float* a = Att + t * 4;
  a[0] = e0 * inv * coef[0]; a[1] = e1 * inv * coef[1];
  a[2] = e2 * inv * coef[2]; a[3] = e3 * inv * coef[3];
}

// res + concat with broadcast gp -> Cat (Bn, P, 2*CO)
__global__ __launch_bounds__(256) void lu2_apply_kernel(
    const bf16* __restrict__ Og, const float* __restrict__ Att,
    const bf16* __restrict__ Gp, bf16* __restrict__ Cat,
    int Bn, int HD, int WD, int HS, int WS, int CO) {
  const int p = blockIdx.x;
  const int b = blockIdx.y;
  const int P = HD * WD;
  int idx[4]; float coef[4];
  lu2_geom(p, WD, HS, WS, (float)HS / HD, (float)WS / WD, idx, coef);
  const float* a = Att + ((size_t)b * P + p) * 4;
  float a0 = a[0], a1 = a[1], a2 = a[2], a3 = a[3];
  const bf16* r0 = Og + ((size_t)b * HS * WS + idx[0]) * CO;
  const bf16* r1 = Og + ((size_t)b * HS * WS + idx[1]) * CO;
  const bf16* r2 = Og + ((size_t)b * HS * WS + idx[2]) * CO;
  const bf16* r3 = Og + ((size_t)b * HS * WS + idx[3]) * CO;
  bf16* dst = Cat + ((size_t)b * P + p) * (size_t)(2 * CO);
  for (int c = threadIdx.x; c < CO; c += blockDim.x) {
    float v = b2f(r0[c]) * a0 + b2f(r1[c]) * a1 + b2f(r2[c]) * a2 + b2f(r3[c]) * a3;
    dst[c] = f2b(v);
    dst[CO + c] = Gp[(size_t)b * CO + c];
  }
}

// ---------------------------------------------------------------------------
// Host orchestration
// ---------------------------------------------------------------------------
static inline unsigned gblk(size_t tot) { return (unsigned)((tot + 255) / 256); }
static inline int ilog2(int x) { int l = 0; while ((1 << l) < x) ++l; return l; }

extern "C" void kernel_launch(void* const* d_in, const int* in_sizes, int n_in,
                              void* d_out, int out_size, void* d_ws, size_t ws_size,
                              hipStream_t stream) {
  (void)in_sizes; (void)n_in; (void)out_size; (void)ws_size;
  const int Bn = 2;

  const float* c1  = (const float*)d_in[0];   // (2,256,128,128)
  const float* c2  = (const float*)d_in[1];   // (2,512,64,64)
  const float* c3  = (const float*)d_in[2];   // (2,1024,32,32)
  const float* c4  = (const float*)d_in[3];   // (2,2048,16,16)
  const float* c20 = (const float*)d_in[4];   // (2,512,64,64)
  // d_in[5]=c30, d_in[6]=c40 : unused by the reference forward()

  int pi = 7;
  auto PP = [&]() -> const float* { return (const float*)d_in[pi++]; };
  const float *ctx4_w0 = PP(), *ctx4_s0 = PP(), *ctx4_b0 = PP();
  const float *ctx4_w1 = PP(), *ctx4_s1 = PP(), *ctx4_b1 = PP();
  const float *ctx3_w0 = PP(), *ctx3_s0 = PP(), *ctx3_b0 = PP();
  const float *ctx3_w1 = PP(), *ctx3_s1 = PP(), *ctx3_b1 = PP();
  const float *ctx2_w0 = PP(), *ctx2_s0 = PP(), *ctx2_b0 = PP();
  const float *ctx2_w1 = PP(), *ctx2_s1 = PP(), *ctx2_b1 = PP();
  const float *prj4_w = PP(), *prj4_s = PP(), *prj4_b = PP();
  const float *prj3_w = PP(), *prj3_s = PP(), *prj3_b = PP();
  const float *lu4_cn_w = PP(), *lu4_cn_s = PP(), *lu4_cn_b = PP();
  const float *lu4_pr_w = PP(), *lu4_pr_s = PP(), *lu4_pr_b = PP();
  const float *lu4_rf_w = PP(), *lu4_rf_s = PP(), *lu4_rf_b = PP();
  const float *lu4_p2_w = PP(), *lu4_p2_s = PP(), *lu4_p2_b = PP();
  const float *lu3_cn_w = PP(), *lu3_cn_s = PP(), *lu3_cn_b = PP();
  const float *lu3_pr_w = PP(), *lu3_pr_s = PP(), *lu3_pr_b = PP();
  const float *lu3_rf_w = PP(), *lu3_rf_s = PP(), *lu3_rf_b = PP();
  const float *lu3_p2_w = PP(), *lu3_p2_s = PP(), *lu3_p2_b = PP();
  const float *prj_w = PP(), *prj_s = PP(), *prj_b = PP();
  const float *gap_w = PP(), *gap_s = PP(), *gap_b = PP();
  const float *se_w = PP(), *se_b = PP();
  const float *pam_q_w = PP(), *pam_q_b = PP();
  const float *pam_k_w = PP(), *pam_k_b = PP();
  const float *pam_g_w = PP(), *pam_g_b = PP();
  const float *lu2_r1_w = PP(), *lu2_r1_b = PP();
  const float *lu2_r2_w = PP(), *lu2_r2_b = PP();
  const float *c6_w = PP(), *c6_b = PP();

  // bump allocator over workspace
  char* base = (char*)d_ws;
  size_t off = 0;
  auto alloc = [&](size_t bytes) -> char* {
    char* r = base + off;
    off = (off + bytes + 255) & ~(size_t)255;
    return r;
  };
  auto abf = [&](size_t elems) -> bf16* { return (bf16*)alloc(elems * sizeof(bf16)); };
  auto af32 = [&](size_t elems) -> float* { return (float*)alloc(elems * sizeof(float)); };

  auto cvtX = [&](const float* src, int C, int H, int W) -> bf16* {
    bf16* dst = abf((size_t)Bn * C * H * W);
    size_t tot = (size_t)Bn * C * H * W;
    nchw2nhwc_kernel<<<gblk(tot), 256, 0, stream>>>(src, dst, Bn, C, H, W);
    return dst;
  };
  auto cvtW = [&](const float* src, int O, int T, int I) -> bf16* {
    bf16* dst = abf((size_t)O * T * I);
    size_t tot = (size_t)O * T * I;
    wcvt_kernel<<<gblk(tot), 256, 0, stream>>>(src, dst, O, T, I);
    return dst;
  };
  auto conv = [&](const bf16* X, int H, int W, int Cin, const bf16* Wt, int Cout,
                  int ks, int dil, const float* s, const float* b, int act,
                  bf16* Yb, float* Yf) {
    dim3 g((Bn * H * W + 31) / 32, (Cout + 127) / 128, 1);
    conv_wmma_kernel<<<g, 256, 0, stream>>>(X, Wt, s, b, Yb, Yf, Bn, H, W, Cin, Cout,
                                            ks, dil, act, ilog2(H * W), ilog2(W));
  };
  auto gemm = [&](const bf16* A, const bf16* Bm, float* Cf, bf16* Cb,
                  int M, int N, int K, int bkn, size_t sA, size_t sB, size_t sC) {
    dim3 g((M + 31) / 32, (N + 127) / 128, Bn);
    gemm_wmma_kernel<<<g, 256, 0, stream>>>(A, Bm, Cf, Cb, M, N, K, bkn,
                                            (unsigned long long)sA,
                                            (unsigned long long)sB,
                                            (unsigned long long)sC);
  };
  auto resize = [&](const bf16* src, int IH, int IW, int C, int OH, int OW,
                    bf16* dst, int Cd, int coff) {
    size_t tot = (size_t)Bn * OH * OW * C;
    resize_ac_kernel<<<gblk(tot), 256, 0, stream>>>(src, dst, Bn, IH, IW, C, OH, OW, Cd, coff);
  };
  auto copyc = [&](const bf16* src, bf16* dst, size_t P, int Cs, int Cd, int coff) {
    size_t tot = P * (size_t)Cs;
    copy_ch_kernel<<<gblk(tot), 256, 0, stream>>>(src, dst, P, Cs, Cd, coff);
  };

  // ---- convert activations to NHWC bf16
  bf16* Xc4  = cvtX(c4, 2048, 16, 16);
  bf16* Xc3  = cvtX(c3, 1024, 32, 32);
  bf16* Xc2  = cvtX(c2, 512, 64, 64);
  bf16* Xc1  = cvtX(c1, 256, 128, 128);
  bf16* Xc20 = cvtX(c20, 512, 64, 64);

  // ---- convert weights to (O, taps, I) bf16 (stays resident in 192MB L2)
  bf16* Wc4_0 = cvtW(ctx4_w0, 512, 1, 2048);
  bf16* Wc4_1 = cvtW(ctx4_w1, 512, 9, 2048);
  bf16* Wc3_0 = cvtW(ctx3_w0, 512, 1, 512);
  bf16* Wc3_1 = cvtW(ctx3_w1, 512, 9, 512);
  bf16* Wc2_0 = cvtW(ctx2_w0, 512, 1, 512);
  bf16* Wc2_1 = cvtW(ctx2_w1, 512, 9, 512);
  bf16* Wprj4 = cvtW(prj4_w, 512, 1, 1024);
  bf16* Wprj3 = cvtW(prj3_w, 512, 1, 1024);
  bf16* Wl4cn = cvtW(lu4_cn_w, 256, 1, 1024);
  bf16* Wl4pr = cvtW(lu4_pr_w, 256, 1, 512);
  bf16* Wl4rf = cvtW(lu4_rf_w, 256, 9, 512);
  bf16* Wl4p2 = cvtW(lu4_p2_w, 512, 1, 256);
  bf16* Wl3cn = cvtW(lu3_cn_w, 256, 1, 512);
  bf16* Wl3pr = cvtW(lu3_pr_w, 256, 1, 512);
  bf16* Wl3rf = cvtW(lu3_rf_w, 256, 9, 512);
  bf16* Wl3p2 = cvtW(lu3_p2_w, 512, 1, 256);
  bf16* Wprj  = cvtW(prj_w, 512, 1, 3072);
  bf16* Wgap  = cvtW(gap_w, 512, 1, 2048);
  bf16* Wse   = cvtW(se_w, 512, 1, 512);
  bf16* Wq    = cvtW(pam_q_w, 64, 1, 512);
  bf16* Wk    = cvtW(pam_k_w, 64, 1, 512);
  bf16* Wg    = cvtW(pam_g_w, 1, 1, 512);
  bf16* Wr1   = cvtW(lu2_r1_w, 32, 1, 256);
  bf16* Wr2   = cvtW(lu2_r2_w, 32, 1, 512);
  bf16* Wc6   = cvtW(c6_w, 19, 1, 1024);

  // ---- ctx4 @16x16 (Cin=2048)
  bf16* f4_0 = abf((size_t)Bn * 256 * 512);
  conv(Xc4, 16, 16, 2048, Wc4_0, 512, 1, 1, ctx4_s0, ctx4_b0, 1, f4_0, nullptr);
  bf16* f4_1 = abf((size_t)Bn * 256 * 512);
  conv(Xc4, 16, 16, 2048, Wc4_1, 512, 3, 8, ctx4_s1, ctx4_b1, 1, f4_1, nullptr);
  bf16* cat4 = abf((size_t)Bn * 256 * 1024);
  copyc(f4_0, cat4, (size_t)Bn * 256, 512, 1024, 0);
  copyc(f4_1, cat4, (size_t)Bn * 256, 512, 1024, 512);
  bf16* p4 = abf((size_t)Bn * 256 * 512);
  conv(cat4, 16, 16, 1024, Wprj4, 512, 1, 1, prj4_s, prj4_b, 1, p4, nullptr);

  // ---- local_up lu4 @32x32
  bf16* l4c1p = abf((size_t)Bn * 1024 * 256);
  conv(Xc3, 32, 32, 1024, Wl4cn, 256, 1, 1, lu4_cn_s, lu4_cn_b, 1, l4c1p, nullptr);
  bf16* l4c2u = abf((size_t)Bn * 1024 * 512);
  resize(p4, 16, 16, 512, 32, 32, l4c2u, 512, 0);
  bf16* l4c2p = abf((size_t)Bn * 1024 * 256);
  conv(l4c2u, 32, 32, 512, Wl4pr, 256, 1, 1, lu4_pr_s, lu4_pr_b, 1, l4c2p, nullptr);
  bf16* l4cat = abf((size_t)Bn * 1024 * 512);
  copyc(l4c1p, l4cat, (size_t)Bn * 1024, 256, 512, 0);
  copyc(l4c2p, l4cat, (size_t)Bn * 1024, 256, 512, 256);
  bf16* l4rf = abf((size_t)Bn * 1024 * 256);
  conv(l4cat, 32, 32, 512, Wl4rf, 256, 3, 1, lu4_rf_s, lu4_rf_b, 1, l4rf, nullptr);
  bf16* l4p2 = abf((size_t)Bn * 1024 * 512);
  conv(l4rf, 32, 32, 256, Wl4p2, 512, 1, 1, lu4_p2_s, lu4_p2_b, 0, l4p2, nullptr);
  bf16* out3 = abf((size_t)Bn * 1024 * 512);
  add_relu_kernel<<<gblk((size_t)Bn * 1024 * 512), 256, 0, stream>>>(
      l4c2u, l4p2, out3, (size_t)Bn * 1024 * 512);

  // ---- ctx3 @32x32
  bf16* f3_0 = abf((size_t)Bn * 1024 * 512);
  conv(out3, 32, 32, 512, Wc3_0, 512, 1, 1, ctx3_s0, ctx3_b0, 1, f3_0, nullptr);
  bf16* f3_1 = abf((size_t)Bn * 1024 * 512);
  conv(out3, 32, 32, 512, Wc3_1, 512, 3, 8, ctx3_s1, ctx3_b1, 1, f3_1, nullptr);
  bf16* cat3 = abf((size_t)Bn * 1024 * 1024);
  copyc(f3_0, cat3, (size_t)Bn * 1024, 512, 1024, 0);
  copyc(f3_1, cat3, (size_t)Bn * 1024, 512, 1024, 512);
  bf16* p3 = abf((size_t)Bn * 1024 * 512);
  conv(cat3, 32, 32, 1024, Wprj3, 512, 1, 1, prj3_s, prj3_b, 1, p3, nullptr);

  // ---- local_up lu3 @64x64
  bf16* l3c1p = abf((size_t)Bn * 4096 * 256);
  conv(Xc2, 64, 64, 512, Wl3cn, 256, 1, 1, lu3_cn_s, lu3_cn_b, 1, l3c1p, nullptr);
  bf16* l3c2u = abf((size_t)Bn * 4096 * 512);
  resize(p3, 32, 32, 512, 64, 64, l3c2u, 512, 0);
  bf16* l3c2p = abf((size_t)Bn * 4096 * 256);
  conv(l3c2u, 64, 64, 512, Wl3pr, 256, 1, 1, lu3_pr_s, lu3_pr_b, 1, l3c2p, nullptr);
  bf16* l3cat = abf((size_t)Bn * 4096 * 512);
  copyc(l3c1p, l3cat, (size_t)Bn * 4096, 256, 512, 0);
  copyc(l3c2p, l3cat, (size_t)Bn * 4096, 256, 512, 256);
  bf16* l3rf = abf((size_t)Bn * 4096 * 256);
  conv(l3cat, 64, 64, 512, Wl3rf, 256, 3, 1, lu3_rf_s, lu3_rf_b, 1, l3rf, nullptr);
  bf16* l3p2 = abf((size_t)Bn * 4096 * 512);
  conv(l3rf, 64, 64, 256, Wl3p2, 512, 1, 1, lu3_p2_s, lu3_p2_b, 0, l3p2, nullptr);
  bf16* out2 = abf((size_t)Bn * 4096 * 512);
  add_relu_kernel<<<gblk((size_t)Bn * 4096 * 512), 256, 0, stream>>>(
      l3c2u, l3p2, out2, (size_t)Bn * 4096 * 512);

  // ---- ctx2 @64x64 (cat2 itself is unused downstream)
  bf16* f2_0 = abf((size_t)Bn * 4096 * 512);
  conv(out2, 64, 64, 512, Wc2_0, 512, 1, 1, ctx2_s0, ctx2_b0, 1, f2_0, nullptr);
  bf16* f2_1 = abf((size_t)Bn * 4096 * 512);
  conv(out2, 64, 64, 512, Wc2_1, 512, 3, 8, ctx2_s1, ctx2_b1, 1, f2_1, nullptr);

  // ---- feats concat (2,64,64,3072) and prj
  bf16* feats = abf((size_t)Bn * 4096 * 3072);
  copyc(f2_0, feats, (size_t)Bn * 4096, 512, 3072, 0);
  copyc(f2_1, feats, (size_t)Bn * 4096, 512, 3072, 512);
  resize(f3_0, 32, 32, 512, 64, 64, feats, 3072, 1024);
  resize(f3_1, 32, 32, 512, 64, 64, feats, 3072, 1536);
  resize(f4_0, 16, 16, 512, 64, 64, feats, 3072, 2048);
  resize(f4_1, 16, 16, 512, 64, 64, feats, 3072, 2560);
  bf16* outb = abf((size_t)Bn * 4096 * 512);
  conv(feats, 64, 64, 3072, Wprj, 512, 1, 1, prj_s, prj_b, 1, outb, nullptr);

  // ---- gp / se
  bf16* gpin = abf((size_t)Bn * 2048);
  gap_mean_kernel<<<gblk(Bn * 2048), 256, 0, stream>>>(c4, gpin, Bn * 2048, 256);
  bf16* gpb = abf((size_t)Bn * 512);
  conv(gpin, 1, 1, 2048, Wgap, 512, 1, 1, gap_s, gap_b, 1, gpb, nullptr);
  bf16* seb = abf((size_t)Bn * 512);
  conv(gpb, 1, 1, 512, Wse, 512, 1, 1, nullptr, se_b, 2, seb, nullptr);
  scale_se_kernel<<<gblk((size_t)Bn * 4096 * 512), 256, 0, stream>>>(
      outb, seb, (size_t)Bn * 4096 * 512, 4096, 512);

  // ---- PAM
  bf16* xp = abf((size_t)Bn * 1024 * 512);
  maxpool2_kernel<<<gblk((size_t)Bn * 1024 * 512), 256, 0, stream>>>(outb, xp, Bn, 64, 64, 512);
  bf16* qb = abf((size_t)Bn * 4096 * 64);
  conv(outb, 64, 64, 512, Wq, 64, 1, 1, nullptr, pam_q_b, 0, qb, nullptr);
  bf16* kb = abf((size_t)Bn * 1024 * 64);
  conv(xp, 32, 32, 512, Wk, 64, 1, 1, nullptr, pam_k_b, 0, kb, nullptr);
  float* sc = af32((size_t)Bn * 4096 * 1024);
  gemm(qb, kb, sc, nullptr, 4096, 1024, 64, 0,
       (size_t)4096 * 64, (size_t)1024 * 64, (size_t)4096 * 1024);
  bf16* att = abf((size_t)Bn * 4096 * 1024);
  softmax_rows_kernel<<<Bn * 4096, 256, 0, stream>>>(sc, att, 1024);
  bf16* core = abf((size_t)Bn * 4096 * 512);
  gemm(att, xp, nullptr, core, 4096, 512, 1024, 1,
       (size_t)4096 * 1024, (size_t)1024 * 512, (size_t)4096 * 512);
  bf16* gb = abf((size_t)Bn * 4096);
  conv(outb, 64, 64, 512, Wg, 1, 1, 1, nullptr, pam_g_b, 2, gb, nullptr);
  bf16* pamo = abf((size_t)Bn * 4096 * 512);
  pam_mix_kernel<<<gblk((size_t)Bn * 4096 * 512), 256, 0, stream>>>(
      core, outb, gb, pamo, (size_t)Bn * 4096 * 512, 512);

  // ---- local_up2 + final concat
  bf16* q2 = abf((size_t)Bn * 16384 * 32);
  conv(Xc1, 128, 128, 256, Wr1, 32, 1, 1, nullptr, lu2_r1_b, 0, q2, nullptr);
  bf16* k2 = abf((size_t)Bn * 4096 * 32);
  conv(Xc20, 64, 64, 512, Wr2, 32, 1, 1, nullptr, lu2_r2_b, 0, k2, nullptr);
  float* att2 = af32((size_t)Bn * 16384 * 4);
  lu2_att_kernel<<<gblk((size_t)Bn * 16384), 256, 0, stream>>>(
      q2, k2, att2, Bn, 128, 128, 64, 64, 32);
  bf16* catf = abf((size_t)Bn * 16384 * 1024);
  lu2_apply_kernel<<<dim3(16384, Bn, 1), 256, 0, stream>>>(
      pamo, att2, gpb, catf, Bn, 128, 128, 64, 64, 512);

  // ---- final conv c6 -> f32 NCHW output (2,19,128,128)
  conv(catf, 128, 128, 1024, Wc6, 19, 1, 1, nullptr, c6_b, 0, nullptr, (float*)d_out);
}
